// TFMPTF_46127948759232
// MI455X (gfx1250) — matrix operand order repaired and checked by reading, not compiled, not run on hardware
//
#include <hip/hip_runtime.h>
#include <math.h>

typedef __attribute__((ext_vector_type(16))) _Float16 v16h;
typedef __attribute__((ext_vector_type(8)))  _Float16 v8h;
typedef __attribute__((ext_vector_type(8)))  float    v8f;

#define B_    16
#define T_    4096
#define D_    512
#define G_    32
#define K_    8
#define SPG   16          // D_/G_
#define MP    24
#define NWIN  (T_ - 4 + 1)     // 4093 windows
#define NTRANS (NWIN - 1)      // 4092 transitions per mode
#define TOTCNT (K_ * NTRANS)   // 32736 counts per (b,g)

// ---- workspace layout (in floats) ----
#define WS_TAPS   ((size_t)0)                          // K_*T_        = 32768
#define WS_STATS  (WS_TAPS + (size_t)K_*T_)            // B*G*K*2      = 8192
#define WS_POOLED (WS_STATS + (size_t)B_*G_*K_*2)      // B*G*T        = 2097152
#define WS_MODES  (WS_POOLED + (size_t)B_*G_*T_)       // B*G*K*T      = 16777216
// total ~18.9M floats ~ 75.7 MB

// ------------------------------------------------------------------
// K1: group pooling.  hs[B,T,D] -> pooled[B,G,T], mean over 16 chans.
// ------------------------------------------------------------------
__global__ __launch_bounds__(128) void k_pool(const float* __restrict__ hs,
                                              float* __restrict__ pooled) {
  int bt  = blockIdx.x;            // b*T_ + t
  int tid = threadIdx.x;
  __shared__ float s[128];
  const float4* row = (const float4*)(hs + (size_t)bt * D_);
  float4 v = row[tid];
  s[tid] = v.x + v.y + v.z + v.w;
  __syncthreads();
  if (tid < G_) {
    float sum = s[tid*4] + s[tid*4+1] + s[tid*4+2] + s[tid*4+3];
    int b = bt / T_, t = bt - b * T_;
    pooled[((size_t)b * G_ + tid) * T_ + t] = sum * (1.0f / SPG);
  }
}

// ------------------------------------------------------------------
// K2: exact real filter taps  g_k[d] = (1/T) sum_j H_k(f_j) cos(2*pi*j*d/T)
// Angle reduced exactly via (j*d) & (T-1) so cosf stays accurate.
// ------------------------------------------------------------------
__global__ __launch_bounds__(256) void k_taps(float* __restrict__ taps) {
  int k   = blockIdx.x;            // 0..K_-1
  int tid = threadIdx.x;
  __shared__ float ctab[T_];
  __shared__ float hk[T_];
  const float TWO_PI = 6.28318530717958647692f;
  float ck = -0.5f + (float)k * (1.0f / 7.0f);       // linspace(-0.5,0.5,8)
  for (int j = tid; j < T_; j += 256) {
    ctab[j] = cosf(TWO_PI * (float)j * (1.0f / T_));
    float f = (j < T_/2 ? (float)j : (float)(j - T_)) * (1.0f / T_);
    float z = fabsf(f - ck) * 8.0f;                  // dist / bandwidth(0.125)
    hk[j] = expf(-0.5f * z * z);
  }
  __syncthreads();
  for (int d = tid; d < T_; d += 256) {
    float acc = 0.f;
    for (int j = 0; j < T_; ++j)
      acc = fmaf(hk[j], ctab[(j * d) & (T_ - 1)], acc);
    taps[(size_t)k * T_ + d] = acc * (1.0f / T_);
  }
}

// ------------------------------------------------------------------
// K3: exact circular convolution as WMMA GEMM against an on-the-fly
// circulant.  modes_k = X . G_k with G_k[tau,t] = g_k[(t-tau) mod T].
// One block per (k, 16-row bg tile); X staged f16 in LDS (128 KB),
// reversed tap table grev[d] = g[(-d) mod T] padded by 16 so every
// B fragment is 16 ascending contiguous halfs.
// ------------------------------------------------------------------
__global__ __launch_bounds__(256) void k_conv(const float* __restrict__ pooled,
                                              const float* __restrict__ taps,
                                              float* __restrict__ modes) {
  int bx  = blockIdx.x;
  int k   = bx >> 5;               // 0..7
  int bgT = bx & 31;               // 0..31  (16 bg rows per tile)
  int tid = threadIdx.x;
  __shared__ _Float16 Xl[16][T_];          // 128 KB
  __shared__ _Float16 grev[T_ + 16];       // 8 KB (+pad for wrap-free reads)

  for (int i = tid; i < 16 * T_; i += 256) {
    int m = i >> 12, t = i & (T_ - 1);
    Xl[m][t] = (_Float16)pooled[((size_t)(bgT * 16 + m)) * T_ + t];
  }
  const float* gk = taps + (size_t)k * T_;
  for (int i = tid; i < T_ + 16; i += 256)
    grev[i] = (_Float16)gk[(T_ - i) & (T_ - 1)];
  __syncthreads();

  int wave = tid >> 5, lane = tid & 31;
  int n = lane & 15, hi = lane >> 4;

  for (int tt = wave; tt < T_ / 16; tt += 8) {   // 32 t-tiles per wave
    int t0 = tt * 16;
    v8f c = {};
    for (int tau = 0; tau < T_; tau += 32) {
      // A fragment: lane<16 -> K {0..15},
      // lane>=16 -> K {16..31}; row M = lane&15.
      v8h alo = *(const v8h*)&Xl[n][tau + hi * 16];
      v8h ahi = *(const v8h*)&Xl[n][tau + hi * 16 + 8];
      v16h A = __builtin_shufflevector(alo, ahi,
                 0,1,2,3,4,5,6,7,8,9,10,11,12,13,14,15);
      // B fragment: col N = lane&15, K = e + hi*16.
      // B[e] = g[(t0+n - tau - (e+hi*16)) mod T] = grev[c0 + e]
      int c0 = (tau + hi * 16 - t0 - n + 2 * T_) & (T_ - 1);
      v16h Bm;
      #pragma unroll
      for (int e = 0; e < 16; ++e) Bm[e] = grev[c0 + e];
      c = __builtin_amdgcn_wmma_f32_16x16x32_f16(false, A, false, Bm,
                                                 (short)0, c, false, false);
    }
    // C layout: lane<16 -> N=lane, vgpr v -> M=v; lane>=16 -> M=8+v.
    #pragma unroll
    for (int v = 0; v < 8; ++v) {
      int bg = bgT * 16 + hi * 8 + v;
      modes[((size_t)bg * K_ + k) * T_ + t0 + n] = c[v];
    }
  }
}

// ------------------------------------------------------------------
// K4: per-(b,g,k) energy mean / unbiased std (ddof=1).
// ------------------------------------------------------------------
__global__ __launch_bounds__(256) void k_stats(const float* __restrict__ modes,
                                               float* __restrict__ stats) {
  int bgk = blockIdx.x;
  int tid = threadIdx.x;
  __shared__ float red[512];
  const float* row = modes + (size_t)bgk * T_;
  float se = 0.f, se2 = 0.f;
  for (int t = tid; t < T_; t += 256) {
    __builtin_prefetch(row + t + 1024, 0, 1);
    float m = row[t];
    float e = m * m;
    se += e; se2 += e * e;
  }
  red[tid] = se; red[256 + tid] = se2;
  __syncthreads();
  for (int ofs = 128; ofs > 0; ofs >>= 1) {
    if (tid < ofs) { red[tid] += red[tid + ofs]; red[256 + tid] += red[256 + tid + ofs]; }
    __syncthreads();
  }
  if (tid == 0) {
    float mean = red[0] * (1.0f / T_);
    float var  = (red[256] - (float)T_ * mean * mean) * (1.0f / (T_ - 1));
    float sd   = fmaxf(sqrtf(fmaxf(var, 0.f)), 1e-8f);
    stats[(size_t)bgk * 2]     = mean;
    stats[(size_t)bgk * 2 + 1] = 1.0f / sd;
  }
}

// ------------------------------------------------------------------
// Stable-argsort Lehmer code for a window of 4 (branch-free, scalar).
// ------------------------------------------------------------------
__device__ __forceinline__ int pid4(float w0, float w1, float w2, float w3) {
  int r0 = (int)(w1 <  w0) + (int)(w2 <  w0) + (int)(w3 <  w0);
  int r1 = (int)(w0 <= w1) + (int)(w2 <  w1) + (int)(w3 <  w1);
  int r2 = (int)(w0 <= w2) + (int)(w1 <= w2) + (int)(w3 <  w2);
  int r3 = (int)(w0 <= w3) + (int)(w1 <= w3) + (int)(w2 <= w3);
  int si0 = 1*(int)(r1==0) + 2*(int)(r2==0) + 3*(int)(r3==0);
  int si1 = 1*(int)(r1==1) + 2*(int)(r2==1) + 3*(int)(r3==1);
  int si2 = 1*(int)(r1==2) + 2*(int)(r2==2) + 3*(int)(r3==2);
  int si3 = 1*(int)(r1==3) + 2*(int)(r2==3) + 3*(int)(r3==3);
  int c0 = (int)(si0 > si1) + (int)(si0 > si2) + (int)(si0 > si3);
  int c1 = (int)(si1 > si2) + (int)(si1 > si3);
  int c2 = (int)(si2 > si3);
  return c0 * 6 + c1 * 2 + c2;
}

// ------------------------------------------------------------------
// K5: ordinal-pattern transition histogram per (b,g) via LDS atomics.
// ------------------------------------------------------------------
__global__ __launch_bounds__(256) void k_hist(const float* __restrict__ modes,
                                              float* __restrict__ out_t) {
  int bg  = blockIdx.x;
  int tid = threadIdx.x;
  __shared__ unsigned bins[MP * MP];
  for (int i = tid; i < MP * MP; i += 256) bins[i] = 0u;
  __syncthreads();
  for (int k = 0; k < K_; ++k) {
    const float* row = modes + ((size_t)bg * K_ + k) * T_;
    for (int t = tid; t < NTRANS; t += 256) {
      __builtin_prefetch(row + t + 512, 0, 1);
      float w0 = row[t], w1 = row[t+1], w2 = row[t+2], w3 = row[t+3], w4 = row[t+4];
      int p0 = pid4(w0, w1, w2, w3);
      int p1 = pid4(w1, w2, w3, w4);
      atomicAdd(&bins[p0 * MP + p1], 1u);
    }
  }
  __syncthreads();
  const float norm = 1.0f / (float)TOTCNT;   // row sum is exactly K_*NTRANS
  for (int i = tid; i < MP * MP; i += 256)
    out_t[(size_t)bg * (MP * MP) + i] = (float)bins[i] * norm;
}

// ------------------------------------------------------------------
// K6: energy-correlation gram via WMMA f32<=f16 16x16x32.
// One wave per (b,g); rows 8..15 zero-padded.
// ------------------------------------------------------------------
struct h4 { _Float16 x, y, z, w; };
__device__ __forceinline__ h4 ne4(float4 v, float mean, float inv) {
  h4 r;
  r.x = (_Float16)((v.x * v.x - mean) * inv);
  r.y = (_Float16)((v.y * v.y - mean) * inv);
  r.z = (_Float16)((v.z * v.z - mean) * inv);
  r.w = (_Float16)((v.w * v.w - mean) * inv);
  return r;
}

__global__ __launch_bounds__(32) void k_gram(const float* __restrict__ modes,
                                             const float* __restrict__ stats,
                                             float* __restrict__ out_f) {
  int bg   = blockIdx.x;
  int lane = threadIdx.x;
  int r    = lane & 15;
  int hi   = lane >> 4;
  bool valid = (r < K_);
  int krow = valid ? r : 0;
  float mean = valid ? stats[((size_t)bg * K_ + r) * 2]     : 0.f;
  float inv  = valid ? stats[((size_t)bg * K_ + r) * 2 + 1] : 0.f;   // 0 => zero row
  const float* row = modes + ((size_t)bg * K_ + krow) * T_;

  v8f c = {};
  for (int t0 = 0; t0 < T_; t0 += 32) {
    const float4* p = (const float4*)(row + t0);
    h4 a0 = ne4(p[hi * 2 + 0], mean, inv);
    h4 a1 = ne4(p[hi * 2 + 1], mean, inv);
    h4 a2 = ne4(p[4 + hi * 2 + 0], mean, inv);
    h4 a3 = ne4(p[4 + hi * 2 + 1], mean, inv);
    h4 b0 = ne4(p[hi * 2 + 0], mean, inv);
    h4 b1 = ne4(p[hi * 2 + 1], mean, inv);
    h4 b2 = ne4(p[4 + hi * 2 + 0], mean, inv);
    h4 b3 = ne4(p[4 + hi * 2 + 1], mean, inv);
    v16h A, Bm;
    A[0]=a0.x;  A[1]=a0.y;  A[2]=a0.z;  A[3]=a0.w;
    A[4]=a1.x;  A[5]=a1.y;  A[6]=a1.z;  A[7]=a1.w;
    A[8]=a2.x;  A[9]=a2.y;  A[10]=a2.z; A[11]=a2.w;
    A[12]=a3.x; A[13]=a3.y; A[14]=a3.z; A[15]=a3.w;
    Bm[0]=b0.x;  Bm[1]=b0.y;  Bm[2]=b0.z;  Bm[3]=b0.w;
    Bm[4]=b1.x;  Bm[5]=b1.y;  Bm[6]=b1.z;  Bm[7]=b1.w;
    Bm[8]=b2.x;  Bm[9]=b2.y;  Bm[10]=b2.z; Bm[11]=b2.w;
    Bm[12]=b3.x; Bm[13]=b3.y; Bm[14]=b3.z; Bm[15]=b3.w;
    c = __builtin_amdgcn_wmma_f32_16x16x32_f16(false, A, false, Bm,
                                               (short)0, c, false, false);
  }
  const float sc = 1.0f / (float)T_;
  if (hi == 0 && lane < K_) {
    #pragma unroll
    for (int k = 0; k < K_ - 1; ++k) {
      if (lane > k) {
        int idx = 7 * k - (k * (k - 1)) / 2 + (lane - k - 1);  // triu order
        out_f[(size_t)bg * 28 + idx] = c[k] * sc;
      }
    }
  }
}

// ------------------------------------------------------------------
extern "C" void kernel_launch(void* const* d_in, const int* in_sizes, int n_in,
                              void* d_out, int out_size, void* d_ws, size_t ws_size,
                              hipStream_t stream) {
  (void)in_sizes; (void)n_in; (void)out_size; (void)ws_size;
  const float* hs = (const float*)d_in[0];
  float* ws     = (float*)d_ws;
  float* taps   = ws + WS_TAPS;
  float* stats  = ws + WS_STATS;
  float* pooled = ws + WS_POOLED;
  float* modes  = ws + WS_MODES;
  float* out_t  = (float*)d_out;                          // [B,G,576]
  float* out_f  = out_t + (size_t)B_ * G_ * (MP * MP);    // [B,G,28]

  k_pool <<<B_ * T_,      128, 0, stream>>>(hs, pooled);
  k_taps <<<K_,           256, 0, stream>>>(taps);
  k_conv <<<K_ * 32,      256, 0, stream>>>(pooled, taps, modes);
  k_stats<<<B_ * G_ * K_, 256, 0, stream>>>(modes, stats);
  k_hist <<<B_ * G_,      256, 0, stream>>>(modes, out_t);
  k_gram <<<B_ * G_,      32,  0, stream>>>(modes, stats, out_f);
}
